// DeformableConv2d_5385888989770
// MI455X (gfx1250) — matrix operand order repaired and checked by reading, not compiled, not run on hardware
//
#include <hip/hip_runtime.h>
#include <hip/hip_bf16.h>
#include <math.h>

// ---------------------------------------------------------------------------
// DCNv2 (deformable conv v2) for MI455X / gfx1250, wave32 + WMMA f16->f32.
// B=8, C=64, H=W=128, OC=64, 3x3, stride=1, pad=1. K=9, CK=C*K=576.
// ---------------------------------------------------------------------------

#define BB   8
#define CC   64
#define HH   128
#define WW   128
#define OCC  64
#define KK   9
#define CK   576          // C*K
#define HW   (HH*WW)      // 16384
#define SROW (CK + 8)     // padded LDS row (halfs) for the sampled tile

typedef _Float16 v16h  __attribute__((ext_vector_type(16)));
typedef _Float16 half8 __attribute__((ext_vector_type(8)));
typedef float    v8f   __attribute__((ext_vector_type(8)));

union V16 { v16h v; half8 h[2]; };

// ---------------------------------------------------------------------------
// Kernel 1: fp32 weight (OC,C,3,3) -> f16 flat [oc][c*9+k]
// ---------------------------------------------------------------------------
__global__ __launch_bounds__(256)
void wcvt_kernel(const float* __restrict__ w, _Float16* __restrict__ wh) {
    int i = blockIdx.x * 256 + threadIdx.x;   // grid sized exactly: 36864/256=144
    wh[i] = (_Float16)w[i];
}

// ---------------------------------------------------------------------------
// Kernel 2: offset/mask conv (27 output channels, 3x3, pad 1) + sigmoid(mask).
// One block = one (b, h, 32-wide w tile). x patch staged in LDS.
// ---------------------------------------------------------------------------
__global__ __launch_bounds__(256)
void offmask_kernel(const float* __restrict__ x,
                    const float* __restrict__ ow, const float* __restrict__ ob,
                    const float* __restrict__ mw, const float* __restrict__ mb,
                    float* __restrict__ offs, float* __restrict__ msk) {
    __shared__ float patch[CC * 3 * 36];      // 64 ch x 3 rows x 34(+2 pad) cols

    const int bi = blockIdx.x;
    const int w0 = (bi & 3) * 32;
    const int h  = (bi >> 2) & (HH - 1);
    const int b  = bi >> 9;
    const int t  = threadIdx.x;

    // Stage input patch rows h-1..h+1, cols w0-1..w0+32 (zero padded).
    for (int i = t; i < CC * 3 * 34; i += 256) {
        int col = i % 34;
        int r   = (i / 34) % 3;
        int c   = i / 102;
        int hy  = h - 1 + r;
        int wx  = w0 - 1 + col;
        float v = 0.0f;
        if ((unsigned)hy < HH && (unsigned)wx < WW)
            v = x[((size_t)(b * CC + c) * HH + hy) * WW + wx];
        patch[(c * 3 + r) * 36 + col] = v;
    }
    __syncthreads();

    // 27 oc x 32 px outputs per block.
    for (int i = t; i < 27 * 32; i += 256) {
        int oc = i >> 5;
        int px = i & 31;
        const float* wp;
        float acc;
        if (oc < 18) { wp = ow + oc * CK;        acc = ob[oc];      }
        else         { wp = mw + (oc - 18) * CK; acc = mb[oc - 18]; }
        for (int c = 0; c < CC; ++c) {
            const float* pr = &patch[c * 3 * 36];
            const float* wr = wp + c * 9;
#pragma unroll
            for (int r = 0; r < 3; ++r)
#pragma unroll
                for (int kw = 0; kw < 3; ++kw)
                    acc += wr[r * 3 + kw] * pr[r * 36 + px + kw];
        }
        size_t pix = ((size_t)h) * WW + (w0 + px);
        if (oc < 18)
            offs[((size_t)(b * 18 + oc)) * HW + pix] = acc;
        else
            msk[((size_t)(b * 9 + (oc - 18))) * HW + pix] = 1.0f / (1.0f + expf(-acc));
    }
}

// ---------------------------------------------------------------------------
// Kernel 3: fused bilinear sampling + WMMA GEMM.
// 128 threads (4 waves). One block = 16 output pixels (row segment) x 64 OC.
// ---------------------------------------------------------------------------
__global__ __launch_bounds__(128)
void deform_kernel(const float* __restrict__ x,
                   const _Float16* __restrict__ wh,
                   const float* __restrict__ bias,
                   const float* __restrict__ offs,
                   const float* __restrict__ msk,
                   float* __restrict__ out) {
    __shared__ _Float16 sT[16 * SROW];        // sampled tile, pixel-major, f16
    __shared__ float    cw[144 * 4];          // corner weights (mask folded in)
    __shared__ int      cidx[144 * 4];        // clamped y*W+x per corner

    const int bi = blockIdx.x;
    const int w0 = (bi & 7) << 4;             // 8 tiles of 16 px per row
    const int h  = (bi >> 3) & (HH - 1);
    const int b  = bi >> 10;
    const int t  = threadIdx.x;

    // ---- Step A: per-(tap,pixel) bilinear metadata (shared across channels)
    for (int e = t; e < 144; e += 128) {
        int tap = e >> 4;
        int px  = e & 15;
        int w   = w0 + px;
        int cy  = tap / 3, cx = tap % 3;
        size_t pix = ((size_t)h) * WW + w;
        float dy = offs[((size_t)(b * 18 + tap * 2    )) * HW + pix];
        float dx = offs[((size_t)(b * 18 + tap * 2 + 1)) * HW + pix];
        float m  = msk [((size_t)(b * 9  + tap        )) * HW + pix];
        float py = (float)(h - 1 + cy) + dy;
        float pxx = (float)(w - 1 + cx) + dx;
        float y0f = floorf(py), x0f = floorf(pxx);
        float fy = py - y0f, fx = pxx - x0f;
        int y0 = (int)y0f, x0 = (int)x0f;
#pragma unroll
        for (int j = 0; j < 4; ++j) {
            int yy = y0 + (j >> 1);
            int xx = x0 + (j & 1);
            float wt = ((j >> 1) ? fy : 1.0f - fy) * ((j & 1) ? fx : 1.0f - fx) * m;
            bool valid = ((unsigned)yy < HH) && ((unsigned)xx < WW);
            int yc = yy < 0 ? 0 : (yy > HH - 1 ? HH - 1 : yy);
            int xc = xx < 0 ? 0 : (xx > WW - 1 ? WW - 1 : xx);
            cw[e * 4 + j]   = valid ? wt : 0.0f;
            cidx[e * 4 + j] = yc * WW + xc;
        }
    }
    __syncthreads();

    // ---- Step B: gather 576 x 16 sampled values into LDS (f16, pixel-major)
    for (int i = t; i < CC * KK * 16; i += 128) {
        int px = i & 15;
        int tp = (i >> 4) % 9;
        int c  = i / 144;
        const float* bx = x + (((size_t)(b * CC + c)) << 14);
        int e = (tp << 4) + px;
        float v = cw[4 * e + 0] * bx[cidx[4 * e + 0]]
                + cw[4 * e + 1] * bx[cidx[4 * e + 1]]
                + cw[4 * e + 2] * bx[cidx[4 * e + 2]]
                + cw[4 * e + 3] * bx[cidx[4 * e + 3]];
        sT[px * SROW + c * 9 + tp] = (_Float16)v;
    }
    __syncthreads();

    // ---- Step C: per-wave 16(OC) x 16(px) tile, K=576 via 18 WMMA ops
    const int wv = t >> 5;                    // wave id 0..3 -> OC slice
    const int ln = t & 31;
    const int g  = ln >> 4;                   // lane half-group
    const int mrow = (wv << 4) + (ln & 15);   // output channel (A-matrix row M)
    const _Float16* wrow = wh + (size_t)mrow * CK;
    const _Float16* srow = &sT[(ln & 15) * SROW];

    v8f acc = {0.f, 0.f, 0.f, 0.f, 0.f, 0.f, 0.f, 0.f};
#pragma unroll
    for (int kb = 0; kb < 18; ++kb) {
        int ko = kb << 5;
        V16 a, bfrag;
        // A (16x32 f16): lanes g=0 hold K in {0..7}u{16..23}, g=1 -> +8
        a.h[0] = *(const half8*)(wrow + ko + 8 * g);
        a.h[1] = *(const half8*)(wrow + ko + 16 + 8 * g);
        // B (32x16 f16): lanes g=0 hold K in {0..7}u{16..23}, g=1 -> +8, col = ln&15
        bfrag.h[0] = *(const half8*)(srow + ko + 8 * g);
        bfrag.h[1] = *(const half8*)(srow + ko + 16 + 8 * g);
        acc = __builtin_amdgcn_wmma_f32_16x16x32_f16(
            /*neg_a=*/false, a.v, /*neg_b=*/false, bfrag.v,
            /*c_mod=*/(short)0, acc, /*reuse_a=*/false, /*reuse_b=*/false);
    }

    // ---- D fragment: VGPR r holds M = 8*g + r; col N = ln&15
    const int pxo = w0 + (ln & 15);
#pragma unroll
    for (int r = 0; r < 8; ++r) {
        int oc = (wv << 4) + (g << 3) + r;
        out[(((size_t)(b * OCC + oc)) * HH + h) * WW + pxo] = acc[r] + bias[oc];
    }
}

// ---------------------------------------------------------------------------
extern "C" void kernel_launch(void* const* d_in, const int* in_sizes, int n_in,
                              void* d_out, int out_size, void* d_ws, size_t ws_size,
                              hipStream_t stream) {
    const float* x        = (const float*)d_in[0];
    const float* weight   = (const float*)d_in[1];
    const float* bias     = (const float*)d_in[2];
    const float* offset_w = (const float*)d_in[3];
    const float* offset_b = (const float*)d_in[4];
    const float* mask_w   = (const float*)d_in[5];
    const float* mask_b   = (const float*)d_in[6];
    float* out = (float*)d_out;

    // Workspace layout (all 16B aligned): offsets | sigmoid(mask) | f16 weights
    float*    offs = (float*)d_ws;                       // 8*18*128*128 f32
    float*    msk  = offs + (size_t)BB * 18 * HW;        // 8*9*128*128  f32
    _Float16* wh   = (_Float16*)(msk + (size_t)BB * 9 * HW);  // 64*576 f16

    wcvt_kernel<<<(OCC * CK) / 256, 256, 0, stream>>>(weight, wh);
    offmask_kernel<<<BB * HH * (WW / 32), 256, 0, stream>>>(
        x, offset_w, offset_b, mask_w, mask_b, offs, msk);
    deform_kernel<<<BB * HH * (WW / 16), 128, 0, stream>>>(
        x, wh, bias, offs, msk, out);
}
